// ContrastiveMagnitudeLoss_52819507806305
// MI455X (gfx1250) — compile-verified, matrix operand order checked by reading, not run on hardware
//
#include <hip/hip_runtime.h>
#include <hip/hip_bf16.h>
#include <math.h>

// ---------------------------------------------------------------------------
// ContrastiveMagnitudeLoss for MI455X (gfx1250, wave32, WMMA)
//
//   P, T : [4096, 768] f32
//   cross = P @ T^T  via bf16 hi/lo split (3x v_wmma_f32_16x16x32_bf16),
//   fused with online log-softmax over each row of -dist/0.1.
//   Per-wave 2x2 tile (32 rows x 32 cols, 4 accumulators) -> 1.33 ds_load
//   per WMMA and 4 independent accumulation chains.
// ---------------------------------------------------------------------------

#define B_ROWS 4096
#define D_DIM  768
#define ROWS_PER_WG 64
#define WG_THREADS 128                      // 4 waves
#define MAIN_GRID (B_ROWS / ROWS_PER_WG)    // 64
#define PANEL_COLS 64
#define PANELS (B_ROWS / PANEL_COLS)        // 64
#define KS 128                              // B K-chunk size
#define KCH (D_DIM / KS)                    // 6 chunks
#define KT_PER_CH (KS / 32)                 // 4 k-tiles per chunk
#define ASTRIDE 776                         // 768 + 8 bf16 pad
#define BSTRIDE 136                         // 128 + 8 bf16 pad

typedef __attribute__((ext_vector_type(16))) __bf16 bf16x16;
typedef __attribute__((ext_vector_type(8)))  __bf16 bf16x8;
typedef __attribute__((ext_vector_type(8)))  float  f32x8;

__device__ __forceinline__ void cvt_split(float x, __bf16& h, __bf16& l) {
    h = (__bf16)x;
    l = (__bf16)(x - (float)h);
}

__device__ __forceinline__ bf16x16 cat8(bf16x8 a, bf16x8 b) {
    return __builtin_shufflevector(a, b, 0,1,2,3,4,5,6,7,8,9,10,11,12,13,14,15);
}

// ---------------------------------------------------------------------------
// Kernel 1: per-row stats.  One wave per row.
// ---------------------------------------------------------------------------
__global__ void cml_row_stats(const float* __restrict__ P,
                              const float* __restrict__ T,
                              float* __restrict__ psq,
                              float* __restrict__ tsq,
                              float* __restrict__ magr) {
    const int wave = threadIdx.x >> 5;
    const int lane = threadIdx.x & 31;
    const int r    = blockIdx.x * 8 + wave;

    const float4* p4 = (const float4*)(P + (size_t)r * D_DIM);
    const float4* t4 = (const float4*)(T + (size_t)r * D_DIM);

    float sp = 0.f, st = 0.f, sl = 0.f, sm = 0.f;
    #pragma unroll
    for (int i = lane; i < D_DIM / 4; i += 32) {
        float4 a = p4[i], b = t4[i];
        sp += a.x * a.x + a.y * a.y + a.z * a.z + a.w * a.w;
        st += b.x * b.x + b.y * b.y + b.z * b.z + b.w * b.w;
        sl += fabsf(a.x - b.x) + fabsf(a.y - b.y) + fabsf(a.z - b.z) + fabsf(a.w - b.w);
        sm += fabsf(b.x) + fabsf(b.y) + fabsf(b.z) + fabsf(b.w);
    }
    #pragma unroll
    for (int m = 16; m >= 1; m >>= 1) {
        sp += __shfl_xor(sp, m, 32);
        st += __shfl_xor(st, m, 32);
        sl += __shfl_xor(sl, m, 32);
        sm += __shfl_xor(sm, m, 32);
    }
    if (lane == 0) {
        psq[r]  = sp;
        tsq[r]  = st;
        magr[r] = sl / sm;
    }
}

// ---------------------------------------------------------------------------
// Kernel 2: fused GEMM + online log-softmax, 2x2 wave tiling.
//   WG: 64 rows, panels of 64 columns. Wave w:
//     stripPair = w&1  -> rows [stripPair*32, +32)
//     colHalf   = w>>1 -> panel cols [colHalf*32, +32)
//   A panel (64 x 768 hi/lo) LDS-resident; B staged in K-chunks of 128.
// ---------------------------------------------------------------------------
__global__ __launch_bounds__(WG_THREADS)
void cml_gemm_softmax(const float* __restrict__ P,
                      const float* __restrict__ T,
                      const float* __restrict__ psq,
                      const float* __restrict__ tsq,
                      float* __restrict__ lossr) {
    extern __shared__ __align__(16) char smem[];
    __bf16* sAhi = (__bf16*)smem;                           // 64 x ASTRIDE
    __bf16* sAlo = sAhi + ROWS_PER_WG * ASTRIDE;
    __bf16* sBhi = sAlo + ROWS_PER_WG * ASTRIDE;            // 64 x BSTRIDE
    __bf16* sBlo = sBhi + PANEL_COLS * BSTRIDE;
    float*  diagL = (float*)(sBlo + PANEL_COLS * BSTRIDE);  // [64]
    float*  partM = diagL + ROWS_PER_WG;                    // [2][64]
    float*  partS = partM + 2 * ROWS_PER_WG;                // [2][64]

    const int tid   = threadIdx.x;
    const int wave  = tid >> 5;
    const int lane  = tid & 31;
    const int half  = lane >> 4;        // 0: lanes 0-15, 1: lanes 16-31
    const int lhalf = lane & 15;
    const int gBase = blockIdx.x * ROWS_PER_WG;

    const int stripPair = wave & 1;     // which 32-row group
    const int colHalf   = wave >> 1;    // which 32-col half of the panel

    // ---- stage A panel (64 rows of P) as bf16 hi/lo into LDS --------------
    union Pack { __bf16 h[4]; uint2 u; };
    for (int vid = tid; vid < ROWS_PER_WG * (D_DIM / 4); vid += WG_THREADS) {
        int r = vid / (D_DIM / 4);
        int k = (vid % (D_DIM / 4)) * 4;
        float4 f = *(const float4*)(P + (size_t)(gBase + r) * D_DIM + k);
        Pack ph, pl;
        cvt_split(f.x, ph.h[0], pl.h[0]);
        cvt_split(f.y, ph.h[1], pl.h[1]);
        cvt_split(f.z, ph.h[2], pl.h[2]);
        cvt_split(f.w, ph.h[3], pl.h[3]);
        *(uint2*)&sAhi[r * ASTRIDE + k] = ph.u;
        *(uint2*)&sAlo[r * ASTRIDE + k] = pl.u;
    }

    // ---- per-wave constants -----------------------------------------------
    const int rowLBase = stripPair * 32;          // LDS row base of this wave
    const int gRowBase = gBase + rowLBase;        // global row base
    const int colLBase = colHalf * 32;            // panel-local col base

    float ps[2][8];
    #pragma unroll
    for (int s = 0; s < 2; ++s)
        #pragma unroll
        for (int v = 0; v < 8; ++v)
            ps[s][v] = psq[gRowBase + s * 16 + half * 8 + v];

    float run_m[2][8], run_s[2][8];
    #pragma unroll
    for (int s = 0; s < 2; ++s)
        #pragma unroll
        for (int v = 0; v < 8; ++v) { run_m[s][v] = -3.0e38f; run_s[s][v] = 0.0f; }

    const int kb = half * 8;   // per-half K sub-chunk offset (A and B layouts)
    const __bf16* aHi0 = sAhi + (rowLBase + lhalf) * ASTRIDE;
    const __bf16* aHi1 = aHi0 + 16 * ASTRIDE;
    const __bf16* aLo0 = sAlo + (rowLBase + lhalf) * ASTRIDE;
    const __bf16* aLo1 = aLo0 + 16 * ASTRIDE;
    const __bf16* bHi0 = sBhi + (colLBase + lhalf) * BSTRIDE;
    const __bf16* bHi1 = bHi0 + 16 * BSTRIDE;
    const __bf16* bLo0 = sBlo + (colLBase + lhalf) * BSTRIDE;
    const __bf16* bLo1 = bLo0 + 16 * BSTRIDE;

    // ---- sweep the 64 column panels ---------------------------------------
    for (int ct = 0; ct < PANELS; ++ct) {
        f32x8 acc[2][2];
        #pragma unroll
        for (int s = 0; s < 2; ++s)
            #pragma unroll
            for (int c = 0; c < 2; ++c)
                acc[s][c] = (f32x8){0.f,0.f,0.f,0.f,0.f,0.f,0.f,0.f};

        for (int kc = 0; kc < KCH; ++kc) {
            __syncthreads();   // prior chunk reads done before overwrite
            // stage B chunk: 64 T-rows x 128 K as bf16 hi/lo
            for (int vid = tid; vid < PANEL_COLS * (KS / 4); vid += WG_THREADS) {
                int r = vid >> 5;            // KS/4 == 32
                int k = (vid & 31) * 4;
                float4 f = *(const float4*)(T + (size_t)(ct * PANEL_COLS + r) * D_DIM
                                              + kc * KS + k);
                Pack ph, pl;
                cvt_split(f.x, ph.h[0], pl.h[0]);
                cvt_split(f.y, ph.h[1], pl.h[1]);
                cvt_split(f.z, ph.h[2], pl.h[2]);
                cvt_split(f.w, ph.h[3], pl.h[3]);
                *(uint2*)&sBhi[r * BSTRIDE + k] = ph.u;
                *(uint2*)&sBlo[r * BSTRIDE + k] = pl.u;
            }
            __syncthreads();

            #pragma unroll
            for (int kt = 0; kt < KT_PER_CH; ++kt) {
                const int kB = kt * 32 + kb;            // within B chunk
                const int kA = kc * KS + kB;            // absolute K in A
                bf16x16 ah0 = cat8(*(const bf16x8*)(aHi0 + kA), *(const bf16x8*)(aHi0 + kA + 16));
                bf16x16 al0 = cat8(*(const bf16x8*)(aLo0 + kA), *(const bf16x8*)(aLo0 + kA + 16));
                bf16x16 ah1 = cat8(*(const bf16x8*)(aHi1 + kA), *(const bf16x8*)(aHi1 + kA + 16));
                bf16x16 al1 = cat8(*(const bf16x8*)(aLo1 + kA), *(const bf16x8*)(aLo1 + kA + 16));
                bf16x16 bh0 = cat8(*(const bf16x8*)(bHi0 + kB), *(const bf16x8*)(bHi0 + kB + 16));
                bf16x16 bl0 = cat8(*(const bf16x8*)(bLo0 + kB), *(const bf16x8*)(bLo0 + kB + 16));
                bf16x16 bh1 = cat8(*(const bf16x8*)(bHi1 + kB), *(const bf16x8*)(bHi1 + kB + 16));
                bf16x16 bl1 = cat8(*(const bf16x8*)(bLo1 + kB), *(const bf16x8*)(bLo1 + kB + 16));

                acc[0][0] = __builtin_amdgcn_wmma_f32_16x16x32_bf16(false, ah0, false, bh0, (short)0, acc[0][0], false, false);
                acc[0][1] = __builtin_amdgcn_wmma_f32_16x16x32_bf16(false, ah0, false, bh1, (short)0, acc[0][1], false, false);
                acc[1][0] = __builtin_amdgcn_wmma_f32_16x16x32_bf16(false, ah1, false, bh0, (short)0, acc[1][0], false, false);
                acc[1][1] = __builtin_amdgcn_wmma_f32_16x16x32_bf16(false, ah1, false, bh1, (short)0, acc[1][1], false, false);
                acc[0][0] = __builtin_amdgcn_wmma_f32_16x16x32_bf16(false, ah0, false, bl0, (short)0, acc[0][0], false, false);
                acc[0][1] = __builtin_amdgcn_wmma_f32_16x16x32_bf16(false, ah0, false, bl1, (short)0, acc[0][1], false, false);
                acc[1][0] = __builtin_amdgcn_wmma_f32_16x16x32_bf16(false, ah1, false, bl0, (short)0, acc[1][0], false, false);
                acc[1][1] = __builtin_amdgcn_wmma_f32_16x16x32_bf16(false, ah1, false, bl1, (short)0, acc[1][1], false, false);
                acc[0][0] = __builtin_amdgcn_wmma_f32_16x16x32_bf16(false, al0, false, bh0, (short)0, acc[0][0], false, false);
                acc[0][1] = __builtin_amdgcn_wmma_f32_16x16x32_bf16(false, al0, false, bh1, (short)0, acc[0][1], false, false);
                acc[1][0] = __builtin_amdgcn_wmma_f32_16x16x32_bf16(false, al1, false, bh0, (short)0, acc[1][0], false, false);
                acc[1][1] = __builtin_amdgcn_wmma_f32_16x16x32_bf16(false, al1, false, bh1, (short)0, acc[1][1], false, false);
            }
        }

        // ---- epilogue: dist -> logit, diag capture, online softmax --------
        const int cBase = ct * PANEL_COLS + colLBase;
        float tq[2];
        tq[0] = tsq[cBase + lhalf];
        tq[1] = tsq[cBase + 16 + lhalf];

        #pragma unroll
        for (int s = 0; s < 2; ++s) {
            #pragma unroll
            for (int c = 0; c < 2; ++c) {
                float lg[8];
                #pragma unroll
                for (int v = 0; v < 8; ++v) {
                    float sq = ps[s][v] + tq[c] - 2.0f * acc[s][c][v];
                    lg[v] = -10.0f * __builtin_sqrtf(fmaxf(sq, 0.0f));  // -dist/0.1
                    if (gRowBase + s * 16 + half * 8 + v == cBase + c * 16 + lhalf)
                        diagL[rowLBase + s * 16 + half * 8 + v] = lg[v];
                }
                #pragma unroll
                for (int v = 0; v < 8; ++v) {
                    float mx = lg[v];
                    #pragma unroll
                    for (int m = 8; m >= 1; m >>= 1) mx = fmaxf(mx, __shfl_xor(mx, m, 32));
                    float e = __expf(lg[v] - mx);
                    #pragma unroll
                    for (int m = 8; m >= 1; m >>= 1) e += __shfl_xor(e, m, 32);
                    float nm = fmaxf(run_m[s][v], mx);
                    run_s[s][v] = run_s[s][v] * __expf(run_m[s][v] - nm) + e * __expf(mx - nm);
                    run_m[s][v] = nm;
                }
            }
        }
    }

    // ---- publish per-column-half partial softmax states -------------------
    if (lhalf == 0) {   // lanes 0 and 16 hold replicated row state
        #pragma unroll
        for (int s = 0; s < 2; ++s)
            #pragma unroll
            for (int v = 0; v < 8; ++v) {
                int lr = rowLBase + s * 16 + half * 8 + v;
                partM[colHalf * ROWS_PER_WG + lr] = run_m[s][v];
                partS[colHalf * ROWS_PER_WG + lr] = run_s[s][v];
            }
    }
    __syncthreads();

    // ---- merge the two column-halves, emit per-row loss -------------------
    if (tid < ROWS_PER_WG) {
        float m0 = partM[tid], m1 = partM[ROWS_PER_WG + tid];
        float mm = fmaxf(m0, m1);
        float ss = partS[tid] * __expf(m0 - mm) + partS[ROWS_PER_WG + tid] * __expf(m1 - mm);
        lossr[gBase + tid] = (mm + __logf(ss)) - diagL[tid];
    }
}

// ---------------------------------------------------------------------------
// Kernel 3: deterministic final reduction + output scalars.
// ---------------------------------------------------------------------------
__global__ void cml_finalize(const float* __restrict__ lossr,
                             const float* __restrict__ magr,
                             float* __restrict__ out) {
    __shared__ float bufc[8], bufm[8];
    const int tid = threadIdx.x;
    float sc = 0.f, sm = 0.f;
    for (int i = tid; i < B_ROWS; i += 256) { sc += lossr[i]; sm += magr[i]; }
    #pragma unroll
    for (int m = 16; m >= 1; m >>= 1) {
        sc += __shfl_xor(sc, m, 32);
        sm += __shfl_xor(sm, m, 32);
    }
    if ((tid & 31) == 0) { bufc[tid >> 5] = sc; bufm[tid >> 5] = sm; }
    __syncthreads();
    if (tid == 0) {
        float tc = 0.f, tm = 0.f;
        #pragma unroll
        for (int w = 0; w < 8; ++w) { tc += bufc[w]; tm += bufm[w]; }
        const float contr = tc / (float)B_ROWS;
        const float mag   = tm / (float)B_ROWS;
        out[0] = 0.5f * contr + 0.5f * mag;   // LOSS_SCALE=1, LAMBD=0.5
        out[1] = contr;
        out[2] = mag;
    }
}

// ---------------------------------------------------------------------------
extern "C" void kernel_launch(void* const* d_in, const int* in_sizes, int n_in,
                              void* d_out, int out_size, void* d_ws, size_t ws_size,
                              hipStream_t stream) {
    const float* P = (const float*)d_in[0];   // predicted [4096,768] f32
    const float* T = (const float*)d_in[1];   // target    [4096,768] f32
    float* out = (float*)d_out;               // 3 scalars

    float* W     = (float*)d_ws;              // 64 KB of f32 scratch
    float* psq   = W;                         // [4096]
    float* tsq   = W + B_ROWS;                // [4096]
    float* magr  = W + 2 * B_ROWS;            // [4096]
    float* lossr = W + 3 * B_ROWS;            // [4096]

    cml_row_stats<<<B_ROWS / 8, 256, 0, stream>>>(P, T, psq, tsq, magr);

    const size_t smem = (size_t)(ROWS_PER_WG * ASTRIDE * 2    // A hi/lo
                               + PANEL_COLS * BSTRIDE * 2)    // B hi/lo
                               * sizeof(__bf16)
                        + 5 * ROWS_PER_WG * sizeof(float);    // diagL + partM/S
    cml_gemm_softmax<<<MAIN_GRID, WG_THREADS, smem, stream>>>(P, T, psq, tsq, lossr);

    cml_finalize<<<1, 256, 0, stream>>>(lossr, magr, out);
}
